// Model_69415261438660
// MI455X (gfx1250) — matrix operand order repaired and checked by reading, not compiled, not run on hardware
//
#include <hip/hip_runtime.h>
#include <math.h>

// ---------------------------------------------------------------------------
// MI455X (gfx1250) implementation.
// Dominant cost: sim = zn @ zn^T (16384x16384x128 = 68.7 GFLOP) fused with
// top-17 selection.  zn (bf16) is 4MB -> L2-resident; compute-bound -> WMMA.
// v3: sim kernel keeps all A fragments (32 query rows) in registers for the
// entire kernel (no per-iteration LDS A reloads), top-k scan reads scores as
// float4.  Steady-state inner loop: 8 global_load_b128 (B) + 8 v_wmma +
// score store/scan only.
// ---------------------------------------------------------------------------

typedef __attribute__((ext_vector_type(16))) __bf16 v16bf;
typedef __attribute__((ext_vector_type(8)))  float  v8f;

#define N_PTS   16384
#define D_IN    64
#define C_DIM   128
#define KP1     17          // k+1 neighbors kept (incl. self)

// eval-mode hard-concrete gate: clamp(sigmoid(l/2)*1.2 - 0.1, 0, 1)
__device__ __forceinline__ float hc_gate(float logit) {
  float s = 1.0f / (1.0f + __expf(-logit * 0.5f));
  float g = s * 1.2f - 0.1f;
  return fminf(fmaxf(g, 0.0f), 1.0f);
}

// A-matrix fragment (16xK bf16, row-major buffer: LDS or global)
// contiguous k order, matching load_B_T: lane<16: M=lane, K=k0..k0+15;
// lane>=16: M=lane-16, K offset +16.
__device__ __forceinline__ v16bf load_A_frag(const __bf16* A, int lane, int k0, int ldk) {
  int row = lane & 15;
  int kb  = k0 + ((lane >> 4) << 4);
  const __bf16* p = A + row * ldk + kb;
  v16bf a;
#pragma unroll
  for (int e = 0; e < 8; ++e) a[e] = p[e];
#pragma unroll
  for (int e = 0; e < 8; ++e) a[8 + e] = p[8 + e];
  return a;
}

// B-matrix (KxN) sourced from a row-major "B^T" buffer BT[Ncols][K]
// (weights pre-transposed to bf16; zn is naturally in this form).
// lanes 0-15 carry K=k0..k0+15 (N=lane), lanes 16-31 carry K=k0+16..k0+31.
__device__ __forceinline__ v16bf load_B_T(const __bf16* BT, int lane, int n0, int k0, int ldk) {
  int col = n0 + (lane & 15);
  int kb  = k0 + ((lane >> 4) << 4);
  const __bf16* p = BT + (size_t)col * ldk + kb;
  v16bf b;
#pragma unroll
  for (int e = 0; e < 16; ++e) b[e] = p[e];
  return b;
}

#define WMMA_BF16(A_, B_, C_) \
  __builtin_amdgcn_wmma_f32_16x16x32_bf16(false, (A_), false, (B_), (short)0, (C_), false, false)

// ---------------------------------------------------------------------------
// Prep kernels
// ---------------------------------------------------------------------------
__global__ void gate_x_kernel(const float* __restrict__ x_raw,
                              const float* __restrict__ col_logit,
                              __bf16* __restrict__ xg) {
  size_t idx = (size_t)blockIdx.x * blockDim.x + threadIdx.x;
  if (idx >= (size_t)N_PTS * D_IN) return;
  int j = (int)(idx & (D_IN - 1));
  xg[idx] = (__bf16)(x_raw[idx] * hc_gate(col_logit[j]));
}

__global__ void transpose_bf16_kernel(const float* __restrict__ W,
                                      __bf16* __restrict__ WT, int R, int Cc) {
  int idx = blockIdx.x * blockDim.x + threadIdx.x;
  if (idx >= R * Cc) return;
  int r = idx / Cc, c = idx - r * Cc;
  WT[(size_t)c * R + r] = (__bf16)W[idx];
}

// ---------------------------------------------------------------------------
// Encoder + TypeVIB: h = (xg@W_enc + b)*tg ; mu = h@W_mu+b ; lv = clip(h@W_lv+b)
// z = mu + eps*exp(0.5 lv) ; zn = z * rsqrt(sum z^2 + 1e-12)
// One block = 16 rows; 8 waves each own a 16x16 column tile; mu/lv chains
// interleave (independent accumulators -> no WMMA RAW stalls).
// ---------------------------------------------------------------------------
__global__ void __launch_bounds__(256) encode_vib_kernel(
    const __bf16* __restrict__ xg, const __bf16* __restrict__ WencT,
    const float* __restrict__ b_enc, const float* __restrict__ type_logit,
    const __bf16* __restrict__ WmuT, const float* __restrict__ b_mu,
    const __bf16* __restrict__ WlvT, const float* __restrict__ b_lv,
    const float* __restrict__ eps,
    float* __restrict__ z, __bf16* __restrict__ zbf, __bf16* __restrict__ znbf) {
  __shared__ __bf16 Ax[16 * D_IN];
  __shared__ __bf16 Ah[16 * C_DIM];
  __shared__ float sumsq[16];
  const int tid = threadIdx.x, lane = tid & 31, wave = tid >> 5;
  const int m0 = blockIdx.x * 16;
  for (int i = tid; i < 16 * D_IN; i += 256)
    Ax[i] = xg[(size_t)(m0 + (i >> 6)) * D_IN + (i & 63)];
  if (tid < 16) sumsq[tid] = 0.0f;
  __syncthreads();

  const int n0  = wave * 16;
  const int col = n0 + (lane & 15);
  const int rb  = (lane >> 4) << 3;

  v8f acc = {};
#pragma unroll
  for (int k0 = 0; k0 < D_IN; k0 += 32) {
    v16bf a = load_A_frag(Ax, lane, k0, D_IN);
    v16bf b = load_B_T(WencT, lane, n0, k0, D_IN);
    acc = WMMA_BF16(a, b, acc);
  }
  const float tg = hc_gate(type_logit[0]);
  const float be = b_enc[col];
#pragma unroll
  for (int r = 0; r < 8; ++r)
    Ah[(rb + r) * C_DIM + col] = (__bf16)((acc[r] + be) * tg);
  __syncthreads();

  v8f am = {}, al = {};
#pragma unroll
  for (int k0 = 0; k0 < C_DIM; k0 += 32) {
    v16bf a  = load_A_frag(Ah, lane, k0, C_DIM);
    v16bf bm = load_B_T(WmuT, lane, n0, k0, C_DIM);
    am = WMMA_BF16(a, bm, am);
    v16bf bl = load_B_T(WlvT, lane, n0, k0, C_DIM);
    al = WMMA_BF16(a, bl, al);
  }
  const float bmu = b_mu[col], blv = b_lv[col];
  float zr[8];
#pragma unroll
  for (int r = 0; r < 8; ++r) {
    int grow = m0 + rb + r;
    float mu  = am[r] + bmu;
    float lvv = fminf(fmaxf(al[r] + blv, -10.0f), 10.0f);
    float zz  = mu + eps[(size_t)grow * C_DIM + col] * __expf(0.5f * lvv);
    zr[r] = zz;
    z[(size_t)grow * C_DIM + col]   = zz;
    zbf[(size_t)grow * C_DIM + col] = (__bf16)zz;
    atomicAdd(&sumsq[rb + r], zz * zz);   // ds_add_f32
  }
  __syncthreads();
#pragma unroll
  for (int r = 0; r < 8; ++r) {
    int grow = m0 + rb + r;
    float inv = rsqrtf(sumsq[rb + r] + 1e-12f);
    znbf[(size_t)grow * C_DIM + col] = (__bf16)(zr[r] * inv);
  }
}

// ---------------------------------------------------------------------------
// Fused similarity + top-17.  Block owns 32 query rows; all 8 A fragments
// (2 M-tiles x 4 k-steps) are loaded from global ONCE into registers and
// reused across all 128 candidate blocks.  Per iteration each of the 8 waves
// owns a 16-candidate tile: B loaded once per k-step and fed to both M-tile
// accumulators (2x L2 reuse, independent WMMA chains -> no RAW hazard).
// 8 threads/row keep strided local top-17 lists in LDS (replace-min),
// scanning scores as float4; final per-row merge of 8 lists.
// ---------------------------------------------------------------------------
__global__ void __launch_bounds__(256) sim_topk_kernel(
    const __bf16* __restrict__ zn, int* __restrict__ topk) {
  __shared__ float  scores[32 * C_DIM];      // 16 KB
  __shared__ float  Lv[256 * KP1];           // 17 KB
  __shared__ int    Li[256 * KP1];           // 17 KB
  const int tid = threadIdx.x, lane = tid & 31, wave = tid >> 5;
  const int m0 = blockIdx.x * 32;

  // query A fragments: resident in VGPRs for the whole kernel (64 VGPRs)
  v16bf a0[4], a1[4];
#pragma unroll
  for (int ks = 0; ks < 4; ++ks) {
    a0[ks] = load_A_frag(zn + (size_t)m0 * C_DIM,        lane, ks * 32, C_DIM);
    a1[ks] = load_A_frag(zn + (size_t)(m0 + 16) * C_DIM, lane, ks * 32, C_DIM);
  }

  float* myv = &Lv[tid * KP1];
  int*   myi = &Li[tid * KP1];
#pragma unroll
  for (int s = 0; s < KP1; ++s) { myv[s] = -INFINITY; myi[s] = 0x7fffffff; }
  float minV = -INFINITY; int minI = 0x7fffffff; int minS = 0;
  __syncthreads();

  const int row = tid >> 3, sub = tid & 7;   // 8 threads per query row
  const int cand0base = wave * 16;
  const int ccol = cand0base + (lane & 15);
  const int rb   = (lane >> 4) << 3;
  for (int cb = 0; cb < N_PTS / C_DIM; ++cb) {
    const int c0 = cb * C_DIM;
    // prefetch next candidate block (speculative, toward WGP$/L2)
    if (cb + 1 < N_PTS / C_DIM) {
      const __bf16* nb = zn + (size_t)(c0 + C_DIM + cand0base + (lane & 15)) * C_DIM;
      __builtin_prefetch((const void*)nb, 0, 1);
    }
    v8f acc0 = {}, acc1 = {};
#pragma unroll
    for (int ks = 0; ks < 4; ++ks) {
      v16bf b = load_B_T(zn, lane, c0 + cand0base, ks * 32, C_DIM);  // shared by both M tiles
      acc0 = WMMA_BF16(a0[ks], b, acc0);     // independent chains: no RAW hazard
      acc1 = WMMA_BF16(a1[ks], b, acc1);
    }
#pragma unroll
    for (int r = 0; r < 8; ++r) {
      scores[(rb + r) * C_DIM + ccol]      = acc0[r];
      scores[(16 + rb + r) * C_DIM + ccol] = acc1[r];
    }
    __syncthreads();

    // each of 8 threads per row scans its 16 contiguous candidates as 4x float4
    const float4* srow = (const float4*)&scores[row * C_DIM + sub * 16];
#pragma unroll
    for (int q = 0; q < 4; ++q) {
      float4 v4 = srow[q];
      float vv[4] = {v4.x, v4.y, v4.z, v4.w};
#pragma unroll
      for (int e = 0; e < 4; ++e) {
        float s  = vv[e];
        int   gi = c0 + sub * 16 + q * 4 + e;
        bool better = (s > minV) || (s == minV && gi < minI);
        if (better) {
          myv[minS] = s; myi[minS] = gi;
          minV = myv[0]; minI = myi[0]; minS = 0;
#pragma unroll
          for (int s2 = 1; s2 < KP1; ++s2) {
            float v2 = myv[s2]; int i2 = myi[s2];
            if ((v2 < minV) || (v2 == minV && i2 > minI)) { minV = v2; minI = i2; minS = s2; }
          }
        }
      }
    }
    __syncthreads();
  }

  // merge: one thread per row selects top-17 of 8*17 entries
  // (tie-break: value desc, then index asc — matches lax.top_k ordering)
  if (sub == 0) {
    float* rv = &Lv[(row * 8) * KP1];
    int*   ri = &Li[(row * 8) * KP1];
    int* outp = &topk[(size_t)(m0 + row) * KP1];
    for (int s = 0; s < KP1; ++s) {
      float bv = -INFINITY; int bi = 0x7fffffff; int bp = 0;
      for (int e = 0; e < 8 * KP1; ++e) {
        float v2 = rv[e]; int i2 = ri[e];
        if ((v2 > bv) || (v2 == bv && i2 < bi)) { bv = v2; bi = i2; bp = e; }
      }
      rv[bp] = -INFINITY; ri[bp] = 0x7fffffff;
      outp[s] = bi;
    }
  }
}

// ---------------------------------------------------------------------------
// Scatter-mean aggregation: edge (i, slot) contributes x[i] to agg[idx[i][slot]]
// when idx != i.  One block per edge, one thread per channel.
// ---------------------------------------------------------------------------
__global__ void scatter_kernel(const int* __restrict__ topk, const float* __restrict__ x,
                               float* __restrict__ agg, float* __restrict__ cnt) {
  const int e = blockIdx.x;
  const int i = e / KP1;
  const int d = topk[e];
  if (d == i) return;
  atomicAdd(&agg[(size_t)d * C_DIM + threadIdx.x], x[(size_t)i * C_DIM + threadIdx.x]);
  if (threadIdx.x == 0) atomicAdd(&cnt[d], 1.0f);
}

__global__ void mean_kernel(const float* __restrict__ agg, const float* __restrict__ cnt,
                            __bf16* __restrict__ meanbf) {
  size_t idx = (size_t)blockIdx.x * blockDim.x + threadIdx.x;
  if (idx >= (size_t)N_PTS * C_DIM) return;
  float c = cnt[idx >> 7];
  meanbf[idx] = (__bf16)(agg[idx] / fmaxf(c, 1.0f));
}

// ---------------------------------------------------------------------------
// Generic fused GEMM: out = act(A1@B1 + [A2@B2] + bias).  K split over two
// accumulators per input so successive WMMAs are independent (hazard-free).
// ---------------------------------------------------------------------------
__global__ void __launch_bounds__(256) gemm_fused_kernel(
    const __bf16* __restrict__ A1, const __bf16* __restrict__ A2,
    const __bf16* __restrict__ B1T, const __bf16* __restrict__ B2T,
    const float* __restrict__ bias, int relu,
    float* __restrict__ outF, __bf16* __restrict__ outBF) {
  __shared__ __bf16 S1[16 * C_DIM];
  __shared__ __bf16 S2[16 * C_DIM];
  const int tid = threadIdx.x, lane = tid & 31, wave = tid >> 5;
  const int m0 = blockIdx.x * 16;
  for (int i = tid; i < 16 * C_DIM; i += 256) {
    size_t g = (size_t)(m0 + (i >> 7)) * C_DIM + (i & 127);
    S1[i] = A1[g];
    if (A2) S2[i] = A2[g];
  }
  __syncthreads();

  const int n0  = wave * 16;
  const int col = n0 + (lane & 15);
  const int rb  = (lane >> 4) << 3;

  v8f accs[2] = {{}, {}};                     // self input: even/odd k-steps
#pragma unroll
  for (int ks = 0; ks < 4; ++ks) {
    const int k0 = ks * 32;
    v16bf a = load_A_frag(S1, lane, k0, C_DIM);
    v16bf b = load_B_T(B1T, lane, n0, k0, C_DIM);
    accs[ks & 1] = WMMA_BF16(a, b, accs[ks & 1]);
  }
  v8f accn[2] = {{}, {}};                     // neighbor input
  if (A2) {
#pragma unroll
    for (int ks = 0; ks < 4; ++ks) {
      const int k0 = ks * 32;
      v16bf a = load_A_frag(S2, lane, k0, C_DIM);
      v16bf b = load_B_T(B2T, lane, n0, k0, C_DIM);
      accn[ks & 1] = WMMA_BF16(a, b, accn[ks & 1]);
    }
  }
  const float bb = bias[col];
#pragma unroll
  for (int r = 0; r < 8; ++r) {
    float v = accs[0][r] + accs[1][r] + bb;
    if (A2) v += accn[0][r] + accn[1][r];
    if (relu) v = fmaxf(v, 0.0f);
    size_t g = (size_t)(m0 + rb + r) * C_DIM + col;
    if (outF)  outF[g]  = v;
    if (outBF) outBF[g] = (__bf16)v;
  }
}

// ---------------------------------------------------------------------------
// Host launch
// ---------------------------------------------------------------------------
extern "C" void kernel_launch(void* const* d_in, const int* in_sizes, int n_in,
                              void* d_out, int out_size, void* d_ws, size_t ws_size,
                              hipStream_t stream) {
  const float* x_raw      = (const float*)d_in[0];
  const float* eps        = (const float*)d_in[1];
  const float* col_logit  = (const float*)d_in[2];
  const float* type_logit = (const float*)d_in[3];
  const float* W_enc      = (const float*)d_in[4];
  const float* b_enc      = (const float*)d_in[5];
  const float* W_mu       = (const float*)d_in[6];
  const float* b_mu       = (const float*)d_in[7];
  const float* W_lv       = (const float*)d_in[8];
  const float* b_lv       = (const float*)d_in[9];
  const float* W_self1    = (const float*)d_in[10];
  const float* W_nbr1     = (const float*)d_in[11];
  const float* b1         = (const float*)d_in[12];
  const float* W_self2    = (const float*)d_in[13];
  const float* W_nbr2     = (const float*)d_in[14];
  const float* b2         = (const float*)d_in[15];
  const float* W_head     = (const float*)d_in[16];
  const float* b_head     = (const float*)d_in[17];
  (void)in_sizes; (void)n_in; (void)out_size; (void)ws_size;

  // workspace layout
  char* base = (char*)d_ws;
  size_t off = 0;
  auto alloc = [&](size_t bytes) -> char* {
    char* p = base + off;
    off = (off + bytes + 255) & ~(size_t)255;
    return p;
  };
  __bf16* xg      = (__bf16*)alloc((size_t)N_PTS * D_IN * 2);
  __bf16* WencT   = (__bf16*)alloc((size_t)C_DIM * D_IN * 2);
  __bf16* WmuT    = (__bf16*)alloc((size_t)C_DIM * C_DIM * 2);
  __bf16* WlvT    = (__bf16*)alloc((size_t)C_DIM * C_DIM * 2);
  __bf16* Ws1T    = (__bf16*)alloc((size_t)C_DIM * C_DIM * 2);
  __bf16* Wn1T    = (__bf16*)alloc((size_t)C_DIM * C_DIM * 2);
  __bf16* Ws2T    = (__bf16*)alloc((size_t)C_DIM * C_DIM * 2);
  __bf16* Wn2T    = (__bf16*)alloc((size_t)C_DIM * C_DIM * 2);
  __bf16* WhT     = (__bf16*)alloc((size_t)C_DIM * C_DIM * 2);
  float*  z_f32   = (float*) alloc((size_t)N_PTS * C_DIM * 4);
  __bf16* z_bf    = (__bf16*)alloc((size_t)N_PTS * C_DIM * 2);
  __bf16* zn_bf   = (__bf16*)alloc((size_t)N_PTS * C_DIM * 2);
  int*    topk    = (int*)   alloc((size_t)N_PTS * KP1 * 4);
  float*  agg     = (float*) alloc((size_t)N_PTS * C_DIM * 4);
  float*  cnt     = (float*) alloc((size_t)N_PTS * 4);
  __bf16* mean_bf = (__bf16*)alloc((size_t)N_PTS * C_DIM * 2);
  float*  h1_f32  = (float*) alloc((size_t)N_PTS * C_DIM * 4);
  __bf16* h1_bf   = (__bf16*)alloc((size_t)N_PTS * C_DIM * 2);
  __bf16* h2_bf   = (__bf16*)alloc((size_t)N_PTS * C_DIM * 2);

  const int NB16 = N_PTS / 16;   // 1024 row-blocks (GEMM kernels)
  const int NB32 = N_PTS / 32;   // 512 row-blocks (sim kernel)

  // 1) gate columns -> bf16 activations; transpose weights -> bf16 [Ncols][K]
  gate_x_kernel<<<(N_PTS * D_IN + 255) / 256, 256, 0, stream>>>(x_raw, col_logit, xg);
  transpose_bf16_kernel<<<(D_IN * C_DIM + 255) / 256, 256, 0, stream>>>(W_enc, WencT, D_IN, C_DIM);
  transpose_bf16_kernel<<<(C_DIM * C_DIM + 255) / 256, 256, 0, stream>>>(W_mu,    WmuT, C_DIM, C_DIM);
  transpose_bf16_kernel<<<(C_DIM * C_DIM + 255) / 256, 256, 0, stream>>>(W_lv,    WlvT, C_DIM, C_DIM);
  transpose_bf16_kernel<<<(C_DIM * C_DIM + 255) / 256, 256, 0, stream>>>(W_self1, Ws1T, C_DIM, C_DIM);
  transpose_bf16_kernel<<<(C_DIM * C_DIM + 255) / 256, 256, 0, stream>>>(W_nbr1,  Wn1T, C_DIM, C_DIM);
  transpose_bf16_kernel<<<(C_DIM * C_DIM + 255) / 256, 256, 0, stream>>>(W_self2, Ws2T, C_DIM, C_DIM);
  transpose_bf16_kernel<<<(C_DIM * C_DIM + 255) / 256, 256, 0, stream>>>(W_nbr2,  Wn2T, C_DIM, C_DIM);
  transpose_bf16_kernel<<<(C_DIM * C_DIM + 255) / 256, 256, 0, stream>>>(W_head,  WhT, C_DIM, C_DIM);

  // 2) encoder + VIB + normalize (WMMA)
  encode_vib_kernel<<<NB16, 256, 0, stream>>>(xg, WencT, b_enc, type_logit,
                                              WmuT, b_mu, WlvT, b_lv, eps,
                                              z_f32, z_bf, zn_bf);

  // 3) fused similarity + top-17 (dominant WMMA workload; zn L2-resident)
  sim_topk_kernel<<<NB32, 256, 0, stream>>>(zn_bf, topk);

  // 4) SAGE layer 1: scatter-mean of z, then relu(z@Ws1 + mean@Wn1 + b1)
  hipMemsetAsync(agg, 0, (size_t)N_PTS * C_DIM * 4, stream);
  hipMemsetAsync(cnt, 0, (size_t)N_PTS * 4, stream);
  scatter_kernel<<<N_PTS * KP1, C_DIM, 0, stream>>>(topk, z_f32, agg, cnt);
  mean_kernel<<<(N_PTS * C_DIM + 255) / 256, 256, 0, stream>>>(agg, cnt, mean_bf);
  gemm_fused_kernel<<<NB16, 256, 0, stream>>>(z_bf, mean_bf, Ws1T, Wn1T, b1, 1, h1_f32, h1_bf);

  // 5) SAGE layer 2
  hipMemsetAsync(agg, 0, (size_t)N_PTS * C_DIM * 4, stream);
  hipMemsetAsync(cnt, 0, (size_t)N_PTS * 4, stream);
  scatter_kernel<<<N_PTS * KP1, C_DIM, 0, stream>>>(topk, h1_f32, agg, cnt);
  mean_kernel<<<(N_PTS * C_DIM + 255) / 256, 256, 0, stream>>>(agg, cnt, mean_bf);
  gemm_fused_kernel<<<NB16, 256, 0, stream>>>(h1_bf, mean_bf, Ws2T, Wn2T, b2, 1, nullptr, h2_bf);

  // 6) head: out = h2 @ W_head + b_head (f32 to d_out)
  gemm_fused_kernel<<<NB16, 256, 0, stream>>>(h2_bf, nullptr, WhT, nullptr, b_head, 0,
                                              (float*)d_out, nullptr);
}
